// TreeMeshGPT_76192719831761
// MI455X (gfx1250) — matrix operand order repaired and checked by reading, not compiled, not run on hardware
//
#include <hip/hip_runtime.h>

typedef __attribute__((ext_vector_type(16))) __bf16 bf16x16;
typedef __attribute__((ext_vector_type(8)))  __bf16 bf16x8;
typedef __attribute__((ext_vector_type(8)))  float  f32x8;
typedef __attribute__((ext_vector_type(4)))  int    i32x4;

#define N_B 2
#define N_S 2048
#define N_D 1024
#define N_H 16
#define N_F 4096
#define N_L 8
#define N_Q 128
#define N_DH 64
#define LN_EPS 1e-5f
#define LDSK 40  // padded row stride (halfwords) to break LDS bank conflicts

#if defined(__HIP_DEVICE_COMPILE__) && __has_builtin(__builtin_amdgcn_global_load_async_to_lds_b128)
#define ASYNC_LDS 1
#else
#define ASYNC_LDS 0
#endif

__device__ __forceinline__ unsigned short f2bf(float f) {
  union { float f; unsigned u; } c; c.f = f;
  unsigned r = c.u + 0x7FFFu + ((c.u >> 16) & 1u);
  return (unsigned short)(r >> 16);
}
__device__ __forceinline__ float bf2f(unsigned short b) {
  union { unsigned u; float f; } c; c.u = ((unsigned)b) << 16;
  return c.f;
}

// 16B global -> LDS copy; async on CDNA5 (GLOBAL_LOAD_ASYNC_TO_LDS_B128, ASYNCcnt)
__device__ __forceinline__ void cp16(const unsigned short* g, unsigned short* l) {
#if ASYNC_LDS
  // Builtin params (from diagnostics): (int4 AS1*, int4 AS3*, imm offset, imm cpol).
  // AS(1) value == generic value for global; AS(3) value == low 32 bits of the
  // generic LDS address (ISA 10.2). inttoptr casts only.
  __builtin_amdgcn_global_load_async_to_lds_b128(
      (__attribute__((address_space(1))) i32x4*)(size_t)g,
      (__attribute__((address_space(3))) i32x4*)(unsigned)(size_t)l, 0, 0);
#else
  *(uint4*)l = *(const uint4*)g;
#endif
}
__device__ __forceinline__ void wait_async() {
#if ASYNC_LDS
#if __has_builtin(__builtin_amdgcn_s_wait_asynccnt)
  __builtin_amdgcn_s_wait_asynccnt(0);
#else
  asm volatile("s_wait_asynccnt 0x0" ::: "memory");
#endif
#endif
}

// A operand: 16 contiguous elements at p (two 8-element chunks at p and p+8)
__device__ __forceinline__ bf16x16 ldopA(const unsigned short* p) {
  bf16x8 lo = *(const bf16x8*)p;
  bf16x8 hi = *(const bf16x8*)(p + 8);
  bf16x16 r;
#pragma unroll
  for (int i = 0; i < 8; ++i) { r[i] = lo[i]; r[i + 8] = hi[i]; }
  return r;
}
// B operand: 16 contiguous elements at p
__device__ __forceinline__ bf16x16 ldopB(const unsigned short* p) {
  bf16x8 lo = *(const bf16x8*)p;
  bf16x8 hi = *(const bf16x8*)(p + 8);
  bf16x16 r;
#pragma unroll
  for (int i = 0; i < 8; ++i) { r[i] = lo[i]; r[i + 8] = hi[i]; }
  return r;
}
__device__ __forceinline__ f32x8 wmma_bf16(bf16x16 a, bf16x16 b, f32x8 c) {
  return __builtin_amdgcn_wmma_f32_16x16x32_bf16(false, a, false, b, (short)0, c,
                                                 false, false);
}

// ---------------- weight cast + transpose: f32 [K][N] -> bf16 [N][K] ----------
__global__ __launch_bounds__(256) void castT_kernel(const float* __restrict__ W,
                                                    unsigned short* __restrict__ Wt,
                                                    int K, int N) {
  long long idx = (long long)blockIdx.x * 256 + threadIdx.x;
  if (idx >= (long long)K * N) return;
  int k = (int)(idx / N);
  int n = (int)(idx % N);
  Wt[(long long)n * K + k] = f2bf(W[idx]);
}

// ---------------- layernorm: f32 row -> bf16 row ------------------------------
__global__ __launch_bounds__(256) void ln_kernel(const float* __restrict__ x,
                                                 const float* __restrict__ g,
                                                 unsigned short* __restrict__ y) {
  int row = blockIdx.x;
  const float* xr = x + (size_t)row * N_D;
  int tid = threadIdx.x;
  float s = 0.f, ss = 0.f;
  for (int i = tid; i < N_D; i += 256) { float v = xr[i]; s += v; ss += v * v; }
  __shared__ float r1[256];
  __shared__ float r2[256];
  r1[tid] = s; r2[tid] = ss;
  __syncthreads();
  for (int st = 128; st > 0; st >>= 1) {
    if (tid < st) { r1[tid] += r1[tid + st]; r2[tid] += r2[tid + st]; }
    __syncthreads();
  }
  float mean = r1[0] * (1.f / N_D);
  float var  = r2[0] * (1.f / N_D) - mean * mean;
  float inv  = rsqrtf(var + LN_EPS);
  for (int i = tid; i < N_D; i += 256)
    y[(size_t)row * N_D + i] = f2bf((xr[i] - mean) * inv * g[i]);
}

// ---------------- GEMM: C[M,N] = A(bf16,MxK) * Bt(bf16,NxK)^T -----------------
// 128x128 WG tile, 32x64 per wave (8 WMMA/k-step), double-buffered async LDS.
__global__ __launch_bounds__(256) void gemm_bf16_kernel(
    const unsigned short* __restrict__ A, int lda,
    const unsigned short* __restrict__ Bt,
    int M, int N, int K,
    const float* __restrict__ bias,
    const float* __restrict__ resid,
    float* __restrict__ Cf,
    unsigned short* __restrict__ Cb,
    int ldc, int relu) {
  __shared__ __attribute__((aligned(16))) unsigned short lA[2][128 * LDSK];
  __shared__ __attribute__((aligned(16))) unsigned short lB[2][128 * LDSK];
  const int tid = threadIdx.x;
  const int n0 = blockIdx.x * 128;
  const int m0 = blockIdx.y * 128;
  const int w    = tid >> 5;
  const int lane = tid & 31;
  const int r    = lane & 15;
  const int hi   = lane >> 4;
  const int wm   = w & 3;   // 4 M-subtiles of 32 rows
  const int wn   = w >> 2;  // 2 N-subtiles of 64 cols
  int rowL[2], kcL[2];
#pragma unroll
  for (int i = 0; i < 2; ++i) {
    int idx = tid + 256 * i;
    rowL[i] = idx >> 2;          // 0..127
    kcL[i]  = (idx & 3) * 8;     // 0,8,16,24
  }
  f32x8 z = {0.f, 0.f, 0.f, 0.f, 0.f, 0.f, 0.f, 0.f};
  f32x8 acc[2][4];
#pragma unroll
  for (int i = 0; i < 2; ++i)
#pragma unroll
    for (int j = 0; j < 4; ++j) acc[i][j] = z;
  const int ldb = K;

  auto loadTiles = [&](int k0, int buf) {
#pragma unroll
    for (int i = 0; i < 2; ++i) {
      const unsigned short* ga = A + (size_t)(m0 + rowL[i]) * lda + k0 + kcL[i];
      cp16(ga, &lA[buf][rowL[i] * LDSK + kcL[i]]);
      int nn = n0 + rowL[i]; if (nn >= N) nn = N - 1;  // clamp; masked at store
      const unsigned short* gb = Bt + (size_t)nn * ldb + k0 + kcL[i];
      cp16(gb, &lB[buf][rowL[i] * LDSK + kcL[i]]);
    }
  };

  loadTiles(0, 0);
  const int nk = K / 32;
  for (int ks = 0; ks < nk; ++ks) {
    const int buf = ks & 1;
    wait_async();
    __syncthreads();
    if (ks + 1 < nk) loadTiles((ks + 1) * 32, buf ^ 1);
    bf16x16 aop[2];
#pragma unroll
    for (int i = 0; i < 2; ++i)
      aop[i] = ldopA(&lA[buf][(32 * wm + 16 * i + r) * LDSK + hi * 16]);
#pragma unroll
    for (int j = 0; j < 4; ++j) {
      bf16x16 bop = ldopB(&lB[buf][(64 * wn + 16 * j + r) * LDSK + hi * 16]);
#pragma unroll
      for (int i = 0; i < 2; ++i) acc[i][j] = wmma_bf16(aop[i], bop, acc[i][j]);
    }
  }
#pragma unroll
  for (int i = 0; i < 2; ++i) {
#pragma unroll
    for (int j = 0; j < 4; ++j) {
#pragma unroll
      for (int rr = 0; rr < 8; ++rr) {
        int row = m0 + 32 * wm + 16 * i + rr + hi * 8;
        int col = n0 + 64 * wn + 16 * j + r;
        if (row < M && col < N) {
          float v = acc[i][j][rr];
          if (bias)  v += bias[col];
          if (relu)  v = v > 0.f ? v : 0.f;
          if (resid) v += resid[(size_t)row * ldc + col];
          if (Cf) Cf[(size_t)row * ldc + col] = v;
          else    Cb[(size_t)row * ldc + col] = f2bf(v);
        }
      }
    }
  }
}

// ---------------- V transpose: qkv -> vt[B,H,DH,S] ----------------------------
__global__ __launch_bounds__(256) void vtrans_kernel(
    const unsigned short* __restrict__ qkv, unsigned short* __restrict__ vt) {
  long long idx = (long long)blockIdx.x * 256 + threadIdx.x;
  const long long tot = (long long)N_B * N_S * N_D;
  if (idx >= tot) return;
  int x = (int)(idx % N_D);
  long long bs = idx / N_D;
  int s = (int)(bs % N_S);
  int b = (int)(bs / N_S);
  int h  = x >> 6;
  int dh = x & 63;
  unsigned short v = qkv[(size_t)bs * (3 * N_D) + 2 * N_D + x];
  vt[((size_t)(b * N_H + h) * N_DH + dh) * N_S + s] = v;
}

// ---------------- flash attention: 1 wave per 32-query tile -------------------
__global__ __launch_bounds__(32) void attn_kernel(
    const unsigned short* __restrict__ qkv,
    const unsigned short* __restrict__ vt,
    unsigned short* __restrict__ o) {
  __shared__ __attribute__((aligned(16))) unsigned short pLDS[16 * LDSK];
  const int nqb = N_S / 32;
  int blk = blockIdx.x;
  int qb  = blk % nqb;
  int t   = blk / nqb;
  int hh  = t % N_H;
  int b   = t / N_H;
  int q0  = qb * 32;
  int lane = threadIdx.x;
  int r  = lane & 15;
  int hi = lane >> 4;
  const int row3 = 3 * N_D;
  // Q operands: qop[qs][c] for two 16-row sub-tiles, two K=32 d-chunks
  bf16x16 qop[2][2];
#pragma unroll
  for (int qs = 0; qs < 2; ++qs) {
    const unsigned short* qrow =
        qkv + (size_t)(b * N_S + q0 + 16 * qs + r) * row3 + hh * N_DH;
#pragma unroll
    for (int c = 0; c < 2; ++c) {
      const unsigned short* p = qrow + 32 * c + hi * 16;
      bf16x8 lo  = *(const bf16x8*)p;
      bf16x8 hi8 = *(const bf16x8*)(p + 8);
#pragma unroll
      for (int i = 0; i < 8; ++i) { qop[qs][c][i] = lo[i]; qop[qs][c][i + 8] = hi8[i]; }
    }
  }
  float m[2][8], l[2][8];
  f32x8 z = {0.f, 0.f, 0.f, 0.f, 0.f, 0.f, 0.f, 0.f};
  f32x8 oacc[2][4];
#pragma unroll
  for (int qs = 0; qs < 2; ++qs) {
#pragma unroll
    for (int i = 0; i < 8; ++i) { m[qs][i] = -3.0e38f; l[qs][i] = 0.f; }
#pragma unroll
    for (int j = 0; j < 4; ++j) oacc[qs][j] = z;
  }
  const float scale = 0.125f;  // 1/sqrt(64)
  int kb_end = (q0 + 31) >> 5;
  for (int kb = 0; kb <= kb_end; ++kb) {
    int k0 = kb * 32;
    // V operands once per k-block, reused by both q sub-tiles
    bf16x16 vop[4];
#pragma unroll
    for (int j = 0; j < 4; ++j)
      vop[j] = ldopB(vt + ((size_t)(b * N_H + hh) * N_DH + 16 * j + r) * N_S +
                     k0 + hi * 16);
    // scores: K operands loaded once per d-chunk, reused by both q sub-tiles
    f32x8 s0[2] = {z, z}, s1[2] = {z, z};
#pragma unroll
    for (int c = 0; c < 2; ++c) {
      bf16x16 kop0 = ldopB(qkv + (size_t)(b * N_S + k0 + r) * row3 + N_D +
                           hh * N_DH + 32 * c + hi * 16);
      bf16x16 kop1 = ldopB(qkv + (size_t)(b * N_S + k0 + 16 + r) * row3 + N_D +
                           hh * N_DH + 32 * c + hi * 16);
#pragma unroll
      for (int qs = 0; qs < 2; ++qs) {
        s0[qs] = wmma_bf16(qop[qs][c], kop0, s0[qs]);
        s1[qs] = wmma_bf16(qop[qs][c], kop1, s1[qs]);
      }
    }
#pragma unroll
    for (int qs = 0; qs < 2; ++qs) {
#pragma unroll
      for (int rr = 0; rr < 8; ++rr) {
        int q = q0 + 16 * qs + rr + hi * 8;
        float v0 = s0[qs][rr] * scale; if (k0 + r > q)      v0 = -1.0e30f;
        float v1 = s1[qs][rr] * scale; if (k0 + 16 + r > q) v1 = -1.0e30f;
        float rm = fmaxf(v0, v1);
#pragma unroll
        for (int d = 1; d < 16; d <<= 1) rm = fmaxf(rm, __shfl_xor(rm, d, 16));
        float mn = fmaxf(m[qs][rr], rm);
        float sc = __expf(m[qs][rr] - mn);
        float p0 = __expf(v0 - mn);
        float p1 = __expf(v1 - mn);
        float ps = p0 + p1;
#pragma unroll
        for (int d = 1; d < 16; d <<= 1) ps += __shfl_xor(ps, d, 16);
        l[qs][rr] = l[qs][rr] * sc + ps;
        m[qs][rr] = mn;
#pragma unroll
        for (int j = 0; j < 4; ++j) oacc[qs][j][rr] *= sc;
        int lrow = (rr + hi * 8) * LDSK;
        pLDS[lrow + r]      = f2bf(p0);
        pLDS[lrow + 16 + r] = f2bf(p1);
      }
      __asm__ volatile("s_wait_dscnt 0x0" ::: "memory");
      bf16x16 pop = ldopA(&pLDS[r * LDSK + hi * 16]);  // C-layout -> A-layout
#pragma unroll
      for (int j = 0; j < 4; ++j) oacc[qs][j] = wmma_bf16(pop, vop[j], oacc[qs][j]);
      __asm__ volatile("s_wait_dscnt 0x0" ::: "memory");  // reads done before reuse
    }
  }
#pragma unroll
  for (int qs = 0; qs < 2; ++qs) {
#pragma unroll
    for (int j = 0; j < 4; ++j) {
#pragma unroll
      for (int rr = 0; rr < 8; ++rr) {
        int q = q0 + 16 * qs + rr + hi * 8;
        int d = 16 * j + r;
        o[(size_t)(b * N_S + q) * N_D + hh * N_DH + d] =
            f2bf(oacc[qs][j][rr] / l[qs][rr]);
      }
    }
  }
}

// ---------------- GEGLU: uv(bf16,[M,2F]) -> g(bf16,[M,F]) ---------------------
__global__ __launch_bounds__(256) void geglu_kernel(
    const unsigned short* __restrict__ uv, unsigned short* __restrict__ gb) {
  long long idx = (long long)blockIdx.x * 256 + threadIdx.x;
  const long long tot = (long long)N_B * N_S * N_F;
  if (idx >= tot) return;
  long long row = idx / N_F;
  int j = (int)(idx % N_F);
  float u  = bf2f(uv[row * (2 * N_F) + j]);
  float gt = bf2f(uv[row * (2 * N_F) + N_F + j]);
  const float c = 0.7978845608028654f;
  float ge = 0.5f * gt * (1.f + tanhf(c * (gt + 0.044715f * gt * gt * gt)));
  gb[idx] = f2bf(u * ge);
}

// ---------------- concat builders for heads 2/3 -------------------------------
__global__ __launch_bounds__(256) void concat2_kernel(
    const unsigned short* __restrict__ hf, const float* __restrict__ e1,
    const int* __restrict__ c1, unsigned short* __restrict__ outb) {
  long long idx = (long long)blockIdx.x * 256 + threadIdx.x;
  const long long tot = (long long)N_B * N_S * 2 * N_D;
  if (idx >= tot) return;
  const int W = 2 * N_D;
  long long row = idx / W;
  int c = (int)(idx % W);
  outb[idx] = (c < N_D) ? hf[row * N_D + c]
                        : f2bf(e1[(size_t)c1[row] * N_D + (c - N_D)]);
}
__global__ __launch_bounds__(256) void concat3_kernel(
    const unsigned short* __restrict__ hf, const float* __restrict__ e1,
    const float* __restrict__ e2, const int* __restrict__ c1,
    const int* __restrict__ c2, unsigned short* __restrict__ outb) {
  long long idx = (long long)blockIdx.x * 256 + threadIdx.x;
  const long long tot = (long long)N_B * N_S * 3 * N_D;
  if (idx >= tot) return;
  const int W = 3 * N_D;
  long long row = idx / W;
  int c = (int)(idx % W);
  unsigned short v;
  if (c < N_D)          v = hf[row * N_D + c];
  else if (c < 2 * N_D) v = f2bf(e1[(size_t)c1[row] * N_D + (c - N_D)]);
  else                  v = f2bf(e2[(size_t)c2[row] * N_D + (c - 2 * N_D)]);
  outb[idx] = v;
}

extern "C" void kernel_launch(void* const* d_in, const int* in_sizes, int n_in,
                              void* d_out, int out_size, void* d_ws, size_t ws_size,
                              hipStream_t stream) {
  (void)in_sizes; (void)n_in; (void)out_size; (void)ws_size;
  const int M = N_B * N_S;  // 4096 rows for every GEMM
  const float* acc_fea = (const float*)d_in[0];
  const int*   c1   = (const int*)d_in[1];
  const int*   c2   = (const int*)d_in[2];
  const float* ln1  = (const float*)d_in[3];
  const float* wqkv = (const float*)d_in[4];
  const float* bqkv = (const float*)d_in[5];
  const float* wo   = (const float*)d_in[6];
  const float* bo   = (const float*)d_in[7];
  const float* ln2  = (const float*)d_in[8];
  const float* w1   = (const float*)d_in[9];
  const float* b1   = (const float*)d_in[10];
  const float* w2   = (const float*)d_in[11];
  const float* b2   = (const float*)d_in[12];
  const float* lnf  = (const float*)d_in[13];
  const float* emb1 = (const float*)d_in[14];
  const float* emb2 = (const float*)d_in[15];
  const float* h1_w1 = (const float*)d_in[16]; const float* h1_b1 = (const float*)d_in[17];
  const float* h1_w2 = (const float*)d_in[18]; const float* h1_b2 = (const float*)d_in[19];
  const float* h1_w3 = (const float*)d_in[20]; const float* h1_b3 = (const float*)d_in[21];
  const float* h2_w1 = (const float*)d_in[22]; const float* h2_b1 = (const float*)d_in[23];
  const float* h2_w2 = (const float*)d_in[24]; const float* h2_b2 = (const float*)d_in[25];
  const float* h2_w3 = (const float*)d_in[26]; const float* h2_b3 = (const float*)d_in[27];
  const float* h3_w1 = (const float*)d_in[28]; const float* h3_b1 = (const float*)d_in[29];
  const float* h3_w2 = (const float*)d_in[30]; const float* h3_b2 = (const float*)d_in[31];
  const float* h3_w3 = (const float*)d_in[32]; const float* h3_b3 = (const float*)d_in[33];
  float* out = (float*)d_out;

  char* ws = (char*)d_ws;
  size_t off = 0;
  auto alloc = [&](size_t bytes) -> void* {
    void* p = ws + off;
    off = (off + bytes + 255) & ~(size_t)255;
    return p;
  };
  unsigned short* wqkvT = (unsigned short*)alloc((size_t)N_L * 3 * N_D * N_D * 2);
  unsigned short* woT   = (unsigned short*)alloc((size_t)N_L * N_D * N_D * 2);
  unsigned short* w1T   = (unsigned short*)alloc((size_t)N_L * 2 * N_F * N_D * 2);
  unsigned short* w2T   = (unsigned short*)alloc((size_t)N_L * N_D * N_F * 2);
  unsigned short* h1w1T = (unsigned short*)alloc((size_t)N_D * N_D * 2);
  unsigned short* h1w2T = (unsigned short*)alloc((size_t)N_D * N_D * 2);
  unsigned short* h1w3T = (unsigned short*)alloc((size_t)(N_Q + 2) * N_D * 2);
  unsigned short* h2w1T = (unsigned short*)alloc((size_t)N_D * 2 * N_D * 2);
  unsigned short* h2w2T = (unsigned short*)alloc((size_t)N_D * N_D * 2);
  unsigned short* h2w3T = (unsigned short*)alloc((size_t)N_Q * N_D * 2);
  unsigned short* h3w1T = (unsigned short*)alloc((size_t)N_D * 3 * N_D * 2);
  unsigned short* h3w2T = (unsigned short*)alloc((size_t)N_D * N_D * 2);
  unsigned short* h3w3T = (unsigned short*)alloc((size_t)N_Q * N_D * 2);
  float*          hbuf  = (float*)alloc((size_t)M * N_D * 4);
  unsigned short* hn    = (unsigned short*)alloc((size_t)M * N_D * 2);
  unsigned short* qkvb  = (unsigned short*)alloc((size_t)M * 3 * N_D * 2);
  unsigned short* vtb   = (unsigned short*)alloc((size_t)M * N_D * 2);
  unsigned short* attnb = (unsigned short*)alloc((size_t)M * N_D * 2);
  unsigned short* uvb   = (unsigned short*)alloc((size_t)M * 2 * N_F * 2);
  unsigned short* gbuf  = (unsigned short*)alloc((size_t)M * N_F * 2);
  unsigned short* hfb   = (unsigned short*)alloc((size_t)M * N_D * 2);
  unsigned short* cat2b = (unsigned short*)alloc((size_t)M * 2 * N_D * 2);
  unsigned short* cat3b = (unsigned short*)alloc((size_t)M * 3 * N_D * 2);
  unsigned short* ta    = (unsigned short*)alloc((size_t)M * N_D * 2);
  unsigned short* tb    = (unsigned short*)alloc((size_t)M * N_D * 2);

  auto castT = [&](const float* src, unsigned short* dst, int K, int N) {
    long long tot = (long long)K * N;
    castT_kernel<<<(unsigned)((tot + 255) / 256), 256, 0, stream>>>(src, dst, K, N);
  };
  auto gemm = [&](const unsigned short* A, int lda, const unsigned short* Bt,
                  int K, int N, const float* bias, const float* resid,
                  float* Cf, unsigned short* Cb, int ldc, int relu) {
    dim3 grid((N + 127) / 128, M / 128);
    gemm_bf16_kernel<<<grid, 256, 0, stream>>>(A, lda, Bt, M, N, K, bias, resid,
                                               Cf, Cb, ldc, relu);
  };

  // --- weight prep: cast f32 [K][N] -> bf16 [N][K] ---
  for (int l = 0; l < N_L; ++l) {
    castT(wqkv + (size_t)l * N_D * 3 * N_D, wqkvT + (size_t)l * 3 * N_D * N_D, N_D, 3 * N_D);
    castT(wo   + (size_t)l * N_D * N_D,     woT   + (size_t)l * N_D * N_D,     N_D, N_D);
    castT(w1   + (size_t)l * N_D * 2 * N_F, w1T   + (size_t)l * 2 * N_F * N_D, N_D, 2 * N_F);
    castT(w2   + (size_t)l * N_F * N_D,     w2T   + (size_t)l * N_D * N_F,     N_F, N_D);
  }
  castT(h1_w1, h1w1T, N_D, N_D);
  castT(h1_w2, h1w2T, N_D, N_D);
  castT(h1_w3, h1w3T, N_D, N_Q + 2);
  castT(h2_w1, h2w1T, 2 * N_D, N_D);
  castT(h2_w2, h2w2T, N_D, N_D);
  castT(h2_w3, h2w3T, N_D, N_Q);
  castT(h3_w1, h3w1T, 3 * N_D, N_D);
  castT(h3_w2, h3w2T, N_D, N_D);
  castT(h3_w3, h3w3T, N_D, N_Q);

  // --- h = acc_fea ---
  (void)hipMemcpyAsync(hbuf, acc_fea, (size_t)M * N_D * sizeof(float),
                       hipMemcpyDeviceToDevice, stream);

  const long long elemsD = (long long)M * N_D;
  // --- decoder layers ---
  for (int l = 0; l < N_L; ++l) {
    ln_kernel<<<M, 256, 0, stream>>>(hbuf, ln1 + (size_t)l * N_D, hn);
    gemm(hn, N_D, wqkvT + (size_t)l * 3 * N_D * N_D, N_D, 3 * N_D,
         bqkv + (size_t)l * 3 * N_D, nullptr, nullptr, qkvb, 3 * N_D, 0);
    vtrans_kernel<<<(unsigned)((elemsD + 255) / 256), 256, 0, stream>>>(qkvb, vtb);
    attn_kernel<<<N_B * N_H * (N_S / 32), 32, 0, stream>>>(qkvb, vtb, attnb);
    gemm(attnb, N_D, woT + (size_t)l * N_D * N_D, N_D, N_D,
         bo + (size_t)l * N_D, hbuf, hbuf, nullptr, N_D, 0);
    ln_kernel<<<M, 256, 0, stream>>>(hbuf, ln2 + (size_t)l * N_D, hn);
    gemm(hn, N_D, w1T + (size_t)l * 2 * N_F * N_D, N_D, 2 * N_F,
         b1 + (size_t)l * 2 * N_F, nullptr, nullptr, uvb, 2 * N_F, 0);
    geglu_kernel<<<(unsigned)(((long long)M * N_F + 255) / 256), 256, 0, stream>>>(uvb, gbuf);
    gemm(gbuf, N_F, w2T + (size_t)l * N_D * N_F, N_F, N_D,
         b2 + (size_t)l * N_D, hbuf, hbuf, nullptr, N_D, 0);
  }
  // --- final layernorm ---
  ln_kernel<<<M, 256, 0, stream>>>(hbuf, lnf, hfb);

  const int OUTW = 3 * N_Q + 2;  // 386
  // --- head 1 ---
  gemm(hfb, N_D, h1w1T, N_D, N_D, h1_b1, nullptr, nullptr, ta, N_D, 1);
  gemm(ta,  N_D, h1w2T, N_D, N_D, h1_b2, nullptr, nullptr, tb, N_D, 1);
  gemm(tb,  N_D, h1w3T, N_D, N_Q + 2, h1_b3, nullptr, out, nullptr, OUTW, 0);
  // --- head 2 ---
  concat2_kernel<<<(unsigned)(((long long)M * 2 * N_D + 255) / 256), 256, 0, stream>>>(
      hfb, emb1, c1, cat2b);
  gemm(cat2b, 2 * N_D, h2w1T, 2 * N_D, N_D, h2_b1, nullptr, nullptr, ta, N_D, 1);
  gemm(ta, N_D, h2w2T, N_D, N_D, h2_b2, nullptr, nullptr, tb, N_D, 1);
  gemm(tb, N_D, h2w3T, N_D, N_Q, h2_b3, nullptr, out + (N_Q + 2), nullptr, OUTW, 0);
  // --- head 3 ---
  concat3_kernel<<<(unsigned)(((long long)M * 3 * N_D + 255) / 256), 256, 0, stream>>>(
      hfb, emb1, emb2, c1, c2, cat3b);
  gemm(cat3b, 3 * N_D, h3w1T, 3 * N_D, N_D, h3_b1, nullptr, nullptr, ta, N_D, 1);
  gemm(ta, N_D, h3w2T, N_D, N_D, h3_b2, nullptr, nullptr, tb, N_D, 1);
  gemm(tb, N_D, h3w3T, N_D, N_Q, h3_b3, nullptr, out + (2 * N_Q + 2), nullptr, OUTW, 0);
}
